// VAE_16853451670151
// MI455X (gfx1250) — compile-verified, matrix operand order checked by reading, not run on hardware
//
#include <hip/hip_runtime.h>
#include <hip/hip_bf16.h>
#include <math.h>

#define B_  2048
#define H_  512
#define T_  32
#define D_  1024

typedef __bf16 bf16_t;
typedef __bf16        v16bf __attribute__((ext_vector_type(16)));
typedef float         v8f   __attribute__((ext_vector_type(8)));
typedef unsigned int  u32x4 __attribute__((ext_vector_type(4)));
typedef int           i32x4 __attribute__((ext_vector_type(4)));

#define LDSR 40   // padded LDS row stride (bf16 elems): 80B rows, 16B-aligned frags

#if __has_builtin(__builtin_amdgcn_global_load_async_to_lds_b128)
#define HAS_ASYNC_LDS 1
#else
#define HAS_ASYNC_LDS 0
#endif

typedef __attribute__((address_space(1))) i32x4 g_i32x4;   // global (prints as __device__)
typedef __attribute__((address_space(3))) i32x4 l_i32x4;   // LDS

template <int N>
__device__ __forceinline__ void wait_async() {
#if __has_builtin(__builtin_amdgcn_s_wait_asynccnt)
  __builtin_amdgcn_s_wait_asynccnt(N);
#else
  asm volatile("s_wait_asynccnt %0" :: "i"(N) : "memory");
#endif
}

#if HAS_ASYNC_LDS
// 16B global -> LDS direct copy, tracked by ASYNCcnt (no VGPR staging)
__device__ __forceinline__ void async_cp16(const bf16_t* g, bf16_t* l) {
  __builtin_amdgcn_global_load_async_to_lds_b128((g_i32x4*)(g), (l_i32x4*)(l), 0, 0);
}
#endif

__device__ __forceinline__ v16bf ld_frag(const bf16_t* p) {
  struct alignas(32) Pack { u32x4 a, b; } t;
  t.a = ((const u32x4*)p)[0];
  t.b = ((const u32x4*)p)[1];
  return __builtin_bit_cast(v16bf, t);
}

// C[M,N](f32) = A[M,K](bf16,row-major) * W[N,K](bf16,row-major)^T + bias
// grid.x = M/64 tiles, grid.y = N/128 tiles, 256 threads (8 waves)
__global__ __launch_bounds__(256)
void gemm_bf16_wmma(const bf16_t* __restrict__ A, int lda,
                    const bf16_t* __restrict__ W, int ldb,
                    float* __restrict__ C, int ldc,
                    const float* __restrict__ bias, int K)
{
#if HAS_ASYNC_LDS
  __shared__ bf16_t As[2][64 * LDSR];
  __shared__ bf16_t Ws[2][128 * LDSR];
#else
  __shared__ bf16_t As[1][64 * LDSR];
  __shared__ bf16_t Ws[1][128 * LDSR];
#endif

  const int tid   = threadIdx.x;
  const int lane  = tid & 31;
  const int wave  = tid >> 5;
  const int waveM = (wave & 1) * 32;
  const int waveN = (wave >> 1) * 32;
  const int blockM = blockIdx.x * 64;
  const int blockN = blockIdx.y * 128;

  const int fr = lane & 15;          // row within 16x16 fragment
  const int kh = (lane >> 4) * 16;   // K-half: lane half h takes k = 16h..16h+15 (contiguous order)

  v8f acc00 = {}, acc01 = {}, acc10 = {}, acc11 = {};

  // per-thread staging coordinates (one 16B chunk for A, two for W)
  const int srow = tid >> 2, scol = (tid & 3) * 8;
  const bf16_t* Ag  = A + (size_t)(blockM + srow) * lda + scol;
  const bf16_t* Wg0 = W + (size_t)(blockN + srow) * ldb + scol;
  const bf16_t* Wg1 = Wg0 + (size_t)64 * ldb;

#if HAS_ASYNC_LDS
  // ---- double-buffered async global->LDS pipeline (ASYNCcnt) ----
  const int nIter = K >> 5;
  {
    // prologue: fill buffer 0 for tile 0
    async_cp16(Ag,  &As[0][srow * LDSR + scol]);
    async_cp16(Wg0, &Ws[0][srow * LDSR + scol]);
    async_cp16(Wg1, &Ws[0][(srow + 64) * LDSR + scol]);
  }
  for (int i = 0; i < nIter; ++i) {
    const int k0 = i << 5;
    if (i + 1 < nIter) {
      const int nb = (i + 1) & 1;
      async_cp16(Ag  + k0 + 32, &As[nb][srow * LDSR + scol]);
      async_cp16(Wg0 + k0 + 32, &Ws[nb][srow * LDSR + scol]);
      async_cp16(Wg1 + k0 + 32, &Ws[nb][(srow + 64) * LDSR + scol]);
      wait_async<3>();   // tile i complete; tile i+1 still in flight
    } else {
      wait_async<0>();
    }
    __syncthreads();     // tile i visible to all waves

    const bf16_t* as = As[i & 1];
    const bf16_t* ws = Ws[i & 1];
    v16bf a0 = ld_frag(as + (waveM      + fr) * LDSR + kh);
    v16bf a1 = ld_frag(as + (waveM + 16 + fr) * LDSR + kh);
    v16bf b0 = ld_frag(ws + (waveN      + fr) * LDSR + kh);
    v16bf b1 = ld_frag(ws + (waveN + 16 + fr) * LDSR + kh);

    acc00 = __builtin_amdgcn_wmma_f32_16x16x32_bf16(false, a0, false, b0, (short)0, acc00, false, false);
    acc01 = __builtin_amdgcn_wmma_f32_16x16x32_bf16(false, a0, false, b1, (short)0, acc01, false, false);
    acc10 = __builtin_amdgcn_wmma_f32_16x16x32_bf16(false, a1, false, b0, (short)0, acc10, false, false);
    acc11 = __builtin_amdgcn_wmma_f32_16x16x32_bf16(false, a1, false, b1, (short)0, acc11, false, false);
    __syncthreads();     // all waves done reading buf before it is refilled
  }
#else
  // ---- fallback: synchronous staging ----
  for (int k0 = 0; k0 < K; k0 += 32) {
    *(u32x4*)(&As[0][srow * LDSR + scol])        = *(const u32x4*)(Ag  + k0);
    *(u32x4*)(&Ws[0][srow * LDSR + scol])        = *(const u32x4*)(Wg0 + k0);
    *(u32x4*)(&Ws[0][(srow + 64) * LDSR + scol]) = *(const u32x4*)(Wg1 + k0);
    if (k0 + 32 < K) {
      __builtin_prefetch(Ag  + k0 + 32, 0, 1);
      __builtin_prefetch(Wg0 + k0 + 32, 0, 1);
    }
    __syncthreads();
    v16bf a0 = ld_frag(&As[0][(waveM      + fr) * LDSR + kh]);
    v16bf a1 = ld_frag(&As[0][(waveM + 16 + fr) * LDSR + kh]);
    v16bf b0 = ld_frag(&Ws[0][(waveN      + fr) * LDSR + kh]);
    v16bf b1 = ld_frag(&Ws[0][(waveN + 16 + fr) * LDSR + kh]);
    acc00 = __builtin_amdgcn_wmma_f32_16x16x32_bf16(false, a0, false, b0, (short)0, acc00, false, false);
    acc01 = __builtin_amdgcn_wmma_f32_16x16x32_bf16(false, a0, false, b1, (short)0, acc01, false, false);
    acc10 = __builtin_amdgcn_wmma_f32_16x16x32_bf16(false, a1, false, b0, (short)0, acc10, false, false);
    acc11 = __builtin_amdgcn_wmma_f32_16x16x32_bf16(false, a1, false, b1, (short)0, acc11, false, false);
    __syncthreads();
  }
#endif

  // C/D layout: VGPR r -> M = r + (lane>>4)*8 ; N = lane & 15
  const int rowBase = blockM + waveM + (lane >> 4) * 8;
  const int colBase = blockN + waveN + (lane & 15);
  const float bias0 = bias ? bias[colBase]      : 0.0f;
  const float bias1 = bias ? bias[colBase + 16] : 0.0f;
#pragma unroll
  for (int r = 0; r < 8; ++r) {
    C[(size_t)(rowBase + r)      * ldc + colBase]      = acc00[r] + bias0;
    C[(size_t)(rowBase + r)      * ldc + colBase + 16] = acc01[r] + bias1;
    C[(size_t)(rowBase + 16 + r) * ldc + colBase]      = acc10[r] + bias0;
    C[(size_t)(rowBase + 16 + r) * ldc + colBase + 16] = acc11[r] + bias1;
  }
}

__device__ __forceinline__ float sigf(float x) { return 1.0f / (1.0f + expf(-x)); }

// gate order i,f,g,o ; writes h (bf16) into act[b][H + j]
__global__ void lstm_pointwise(const float* __restrict__ gates,
                               float* __restrict__ c,
                               bf16_t* __restrict__ act_h, int actStride)
{
  int idx = blockIdx.x * blockDim.x + threadIdx.x;   // B*H
  int b = idx >> 9, j = idx & (H_ - 1);
  const float* g = gates + (size_t)b * (4 * H_);
  float iv = g[j], fv = g[H_ + j], gv = g[2 * H_ + j], ov = g[3 * H_ + j];
  float cn = sigf(fv) * c[idx] + sigf(iv) * tanhf(gv);
  c[idx] = cn;
  act_h[(size_t)b * actStride + j] = (bf16_t)(sigf(ov) * tanhf(cn));
}

// per-row argmax(logit + gumbel), one-hot sample, token gather -> next lstm_in
__global__ __launch_bounds__(256)
void argmax_sample(const float* __restrict__ logits_t, int ldl,
                   const float* __restrict__ gu_t,
                   float* __restrict__ samples_t, int ldsm,
                   float* __restrict__ messages_t, int ldm,
                   int*   __restrict__ words_t,
                   const bf16_t* __restrict__ embT,
                   bf16_t* __restrict__ act_x, int actStride)
{
  __shared__ float sval[256];
  __shared__ int   sidx[256];
  const int b = blockIdx.x, tid = threadIdx.x;
  float best = -__builtin_inff(); int bidx = D_;
  for (int d = tid; d < D_; d += 256) {
    float u = gu_t[(size_t)b * D_ + d];
    float v = logits_t[(size_t)b * ldl + d] + (-logf(-logf(u)));
    if (v > best) { best = v; bidx = d; }
  }
  sval[tid] = best; sidx[tid] = bidx;
  __syncthreads();
  for (int s = 128; s > 0; s >>= 1) {
    if (tid < s) {
      float v2 = sval[tid + s]; int i2 = sidx[tid + s];
      if (v2 > sval[tid] || (v2 == sval[tid] && i2 < sidx[tid])) { sval[tid] = v2; sidx[tid] = i2; }
    }
    __syncthreads();
  }
  const int word = sidx[0];
  for (int d = tid; d < D_; d += 256)
    samples_t[(size_t)b * ldsm + d] = (d == word) ? 1.0f : 0.0f;
  if (tid == 0) { messages_t[(size_t)b * ldm] = (float)word; words_t[(size_t)b * ldm] = word; }
  for (int j = tid; j < H_; j += 256)
    act_x[(size_t)b * actStride + j] = embT[(size_t)word * H_ + j];
}

__global__ void dec_gather(const int* __restrict__ words_t, int ldm,
                           const bf16_t* __restrict__ embT,
                           bf16_t* __restrict__ act_x, int actStride)
{
  int idx = blockIdx.x * blockDim.x + threadIdx.x;   // B*H
  int b = idx >> 9, j = idx & (H_ - 1);
  int w = words_t[(size_t)b * ldm];
  act_x[(size_t)b * actStride + j] = embT[(size_t)w * H_ + j];
}

// Wcat[4H,2H] = [Wih | Whh] as bf16
__global__ void cat_convert(const float* __restrict__ Wih, const float* __restrict__ Whh,
                            bf16_t* __restrict__ Wcat)
{
  size_t idx = (size_t)blockIdx.x * blockDim.x + threadIdx.x;  // 4H * 2H
  int n = (int)(idx >> 10), c = (int)(idx & 1023);
  float v = (c < H_) ? Wih[(size_t)n * H_ + c] : Whh[(size_t)n * H_ + (c - H_)];
  Wcat[idx] = (bf16_t)v;
}

__global__ void convert_f32_bf16(const float* __restrict__ s, bf16_t* __restrict__ d, int n)
{ int i = blockIdx.x * blockDim.x + threadIdx.x; if (i < n) d[i] = (bf16_t)s[i]; }

// embT[d][j] = W_th[j][d] + b_th[j]   (token embedding rows, bias folded in)
__global__ void embT_build(const float* __restrict__ W_th, const float* __restrict__ b_th,
                           bf16_t* __restrict__ embT)
{
  int idx = blockIdx.x * blockDim.x + threadIdx.x;   // D*H
  int d = idx >> 9, j = idx & (H_ - 1);
  embT[idx] = (bf16_t)(W_th[(size_t)j * D_ + d] + b_th[j]);
}

__global__ void vec_add(const float* a, const float* b, float* o, int n)
{ int i = blockIdx.x * blockDim.x + threadIdx.x; if (i < n) o[i] = a[i] + b[i]; }

__global__ void fill_bf16(bf16_t* p, int n)
{ int i = blockIdx.x * blockDim.x + threadIdx.x; if (i < n) p[i] = (bf16_t)0.0f; }

__global__ void copy_f32(const float* s, float* d, int n)
{ int i = blockIdx.x * blockDim.x + threadIdx.x; if (i < n) d[i] = s[i]; }

__global__ void fill_f32(float* p, int n)
{ int i = blockIdx.x * blockDim.x + threadIdx.x; if (i < n) p[i] = 0.0f; }

__global__ void recons_k(const float* __restrict__ eps, const float* __restrict__ mean,
                         const float* __restrict__ lv, float* __restrict__ out, int n)
{ int i = blockIdx.x * blockDim.x + threadIdx.x;
  if (i < n) out[i] = eps[i] * expf(0.5f * lv[i]) + mean[i]; }

extern "C" void kernel_launch(void* const* d_in, const int* in_sizes, int n_in,
                              void* d_out, int out_size, void* d_ws, size_t ws_size,
                              hipStream_t stream)
{
  const float* input   = (const float*)d_in[0];
  const float* gu      = (const float*)d_in[1];
  const float* eps     = (const float*)d_in[2];
  const float* enc_Wih = (const float*)d_in[3];
  const float* enc_Whh = (const float*)d_in[4];
  const float* enc_bih = (const float*)d_in[5];
  const float* enc_bhh = (const float*)d_in[6];
  const float* W_ht    = (const float*)d_in[7];
  const float* b_ht    = (const float*)d_in[8];
  const float* W_th    = (const float*)d_in[9];
  const float* b_th    = (const float*)d_in[10];
  const float* dec_Wih = (const float*)d_in[11];
  const float* dec_Whh = (const float*)d_in[12];
  const float* dec_bih = (const float*)d_in[13];
  const float* dec_bhh = (const float*)d_in[14];
  const float* W_mean  = (const float*)d_in[15];
  const float* b_mean  = (const float*)d_in[16];
  const float* W_lv    = (const float*)d_in[17];
  const float* b_lv    = (const float*)d_in[18];

  float* out_recons   = (float*)d_out;                          // [B,H]
  float* out_samples  = out_recons + (size_t)B_ * H_;           // [B,T,D]
  float* out_logits   = out_samples + (size_t)B_ * T_ * D_;     // [B,T,D]
  float* out_messages = out_logits + (size_t)B_ * T_ * D_;      // [B,T]

  char* wsp = (char*)d_ws;
  auto carve = [&](size_t bytes) -> char* {
    char* p = wsp; wsp += (bytes + 255) & ~(size_t)255; return p;
  };
  bf16_t* enc_Wcat = (bf16_t*)carve((size_t)4 * H_ * 2 * H_ * sizeof(bf16_t));
  bf16_t* dec_Wcat = (bf16_t*)carve((size_t)4 * H_ * 2 * H_ * sizeof(bf16_t));
  bf16_t* Wht_bf   = (bf16_t*)carve((size_t)D_ * H_ * sizeof(bf16_t));
  bf16_t* embT     = (bf16_t*)carve((size_t)D_ * H_ * sizeof(bf16_t));
  bf16_t* Wmean_bf = (bf16_t*)carve((size_t)H_ * H_ * sizeof(bf16_t));
  bf16_t* Wlv_bf   = (bf16_t*)carve((size_t)H_ * H_ * sizeof(bf16_t));
  float*  bias_enc = (float*)carve((size_t)4 * H_ * sizeof(float));
  float*  bias_dec = (float*)carve((size_t)4 * H_ * sizeof(float));
  bf16_t* act_enc  = (bf16_t*)carve((size_t)B_ * 2 * H_ * sizeof(bf16_t)); // [B, x|h]
  bf16_t* act_dec  = (bf16_t*)carve((size_t)B_ * 2 * H_ * sizeof(bf16_t));
  float*  c_enc    = (float*)carve((size_t)B_ * H_ * sizeof(float));
  float*  c_dec    = (float*)carve((size_t)B_ * H_ * sizeof(float));
  float*  gatesb   = (float*)carve((size_t)B_ * 4 * H_ * sizeof(float));
  float*  meanb    = (float*)carve((size_t)B_ * H_ * sizeof(float));
  float*  lvb      = (float*)carve((size_t)B_ * H_ * sizeof(float));
  int*    words    = (int*)carve((size_t)B_ * T_ * sizeof(int));

  const int TPB = 256;
  cat_convert<<<(4 * H_ * 2 * H_) / TPB, TPB, 0, stream>>>(enc_Wih, enc_Whh, enc_Wcat);
  cat_convert<<<(4 * H_ * 2 * H_) / TPB, TPB, 0, stream>>>(dec_Wih, dec_Whh, dec_Wcat);
  convert_f32_bf16<<<(D_ * H_) / TPB, TPB, 0, stream>>>(W_ht, Wht_bf, D_ * H_);
  embT_build<<<(D_ * H_) / TPB, TPB, 0, stream>>>(W_th, b_th, embT);
  convert_f32_bf16<<<(H_ * H_) / TPB, TPB, 0, stream>>>(W_mean, Wmean_bf, H_ * H_);
  convert_f32_bf16<<<(H_ * H_) / TPB, TPB, 0, stream>>>(W_lv, Wlv_bf, H_ * H_);
  vec_add<<<(4 * H_) / TPB, TPB, 0, stream>>>(enc_bih, enc_bhh, bias_enc, 4 * H_);
  vec_add<<<(4 * H_) / TPB, TPB, 0, stream>>>(dec_bih, dec_bhh, bias_dec, 4 * H_);
  fill_bf16<<<(B_ * 2 * H_) / TPB, TPB, 0, stream>>>(act_enc, B_ * 2 * H_);
  fill_bf16<<<(B_ * 2 * H_) / TPB, TPB, 0, stream>>>(act_dec, B_ * 2 * H_);
  copy_f32<<<(B_ * H_) / TPB, TPB, 0, stream>>>(input, c_enc, B_ * H_);
  fill_f32<<<(B_ * H_) / TPB, TPB, 0, stream>>>(c_dec, B_ * H_);

  dim3 gGates(B_ / 64, (4 * H_) / 128);   // 32 x 16
  dim3 gLogit(B_ / 64, D_ / 128);         // 32 x 8
  dim3 gHead (B_ / 64, H_ / 128);         // 32 x 4
  const int gBH = (B_ * H_) / TPB;

  // ---- Encoder (T sequential steps) ----
  for (int t = 0; t < T_; ++t) {
    gemm_bf16_wmma<<<gGates, 256, 0, stream>>>(act_enc, 2 * H_, enc_Wcat, 2 * H_,
                                               gatesb, 4 * H_, bias_enc, 2 * H_);
    lstm_pointwise<<<gBH, TPB, 0, stream>>>(gatesb, c_enc, act_enc + H_, 2 * H_);
    gemm_bf16_wmma<<<gLogit, 256, 0, stream>>>(act_enc + H_, 2 * H_, Wht_bf, H_,
                                               out_logits + (size_t)t * D_, T_ * D_, b_ht, H_);
    argmax_sample<<<B_, 256, 0, stream>>>(out_logits + (size_t)t * D_, T_ * D_,
                                          gu + (size_t)t * B_ * D_,
                                          out_samples + (size_t)t * D_, T_ * D_,
                                          out_messages + t, T_,
                                          words + t,
                                          embT, act_enc, 2 * H_);
  }

  // ---- Decoder (T sequential steps) ----
  for (int t = 0; t < T_; ++t) {
    dec_gather<<<gBH, TPB, 0, stream>>>(words + t, T_, embT, act_dec, 2 * H_);
    gemm_bf16_wmma<<<gGates, 256, 0, stream>>>(act_dec, 2 * H_, dec_Wcat, 2 * H_,
                                               gatesb, 4 * H_, bias_dec, 2 * H_);
    lstm_pointwise<<<gBH, TPB, 0, stream>>>(gatesb, c_dec, act_dec + H_, 2 * H_);
  }

  // ---- Heads + reparameterization ----
  gemm_bf16_wmma<<<gHead, 256, 0, stream>>>(act_dec + H_, 2 * H_, Wmean_bf, H_,
                                            meanb, H_, b_mean, H_);
  gemm_bf16_wmma<<<gHead, 256, 0, stream>>>(act_dec + H_, 2 * H_, Wlv_bf, H_,
                                            lvb, H_, b_lv, H_);
  recons_k<<<gBH, TPB, 0, stream>>>(eps, meanb, lvb, out_recons, B_ * H_);
}
